// SpikeDEGCN_25202868093203
// MI455X (gfx1250) — compile-verified, matrix operand order checked by reading, not run on hardware
//
#include <hip/hip_runtime.h>

typedef __attribute__((ext_vector_type(16))) _Float16 v16h;
typedef __attribute__((ext_vector_type(8)))  float    v8f;
typedef __attribute__((ext_vector_type(4)))  float    v4f;

#define F_DIM 256
#define O_DIM 64
#define T_STEPS 32
#define ROWS_PER_WAVE 16
#define WAVES_PER_BLOCK 8
#define ROWS_PER_BLOCK (ROWS_PER_WAVE * WAVES_PER_BLOCK)   // 128
#define KC_COUNT (F_DIM / 32)   // 8  k-chunks of 32
#define CT_COUNT (O_DIM / 16)   // 4  column tiles of 16

// Fused: Bernoulli encode -> f16 WMMA GEMM -> LIF scan in registers -> softmax.
// u streamed exactly once, non-temporal (1.07 GB => ~46us at 23.3 TB/s HBM);
// inputs re-read from L2 each step; W held in LDS as pre-swizzled B-fragments
// and re-read each step (opaque offset defeats LICM so neither gets hoisted
// into registers: keeps waves <256 VGPRs, no s_set_vgpr_msb, high occupancy).
__global__ __launch_bounds__(256)
void snn_lif_fused(const float* __restrict__ inp,   // [N,F] rates
                   const float* __restrict__ u,     // [T,N,F] noise
                   const float* __restrict__ W,     // [F,O]
                   const float* __restrict__ b,     // [O]
                   float* __restrict__ out,         // [N,O] softmax probs
                   int N)
{
    // W in WMMA B-fragment order: frag(kc,ct) = 32 lanes x 16 halves (32 KB)
    __shared__ _Float16 ldsW[KC_COUNT * CT_COUNT * 32 * 16];

    const int tid = threadIdx.x;
    for (int idx = tid; idx < KC_COUNT * CT_COUNT * 32; idx += blockDim.x) {
        const int kc   = idx >> 7;          // / (CT_COUNT*32)
        const int ct   = (idx >> 5) & 3;
        const int ln   = idx & 31;
        const int n    = ct * 16 + (ln & 15);
        const int kbas = kc * 32 + ((ln >> 4) << 3);
        _Float16* dst  = &ldsW[idx * 16];
#pragma unroll
        for (int h = 0; h < 16; ++h) {
            const int k = kbas + ((h >> 3) << 4) + (h & 7);
            dst[h] = (_Float16)W[k * O_DIM + n];
        }
    }
    __syncthreads();

    const int lane    = tid & 31;
    const int wave    = tid >> 5;
    const int rowBase = blockIdx.x * ROWS_PER_BLOCK + wave * ROWS_PER_WAVE;
    const int rowA    = rowBase + (lane & 15);   // A-matrix row for this lane
    const int koff    = (lane >> 4) << 3;        // 0 or 8 (A layout half-split)

    float bv[CT_COUNT];
#pragma unroll
    for (int ct = 0; ct < CT_COUNT; ++ct) bv[ct] = b[ct * 16 + (lane & 15)];

    v8f v[CT_COUNT], c[CT_COUNT];
#pragma unroll
    for (int ct = 0; ct < CT_COUNT; ++ct)
#pragma unroll
        for (int i = 0; i < 8; ++i) { v[ct][i] = 0.0f; c[ct][i] = 0.0f; }

    const float* inRow = inp + (size_t)rowA * F_DIM;
    const size_t tStride = (size_t)N * F_DIM;

#pragma unroll 1
    for (int t = 0; t < T_STEPS; ++t) {
        // Opaque zero: blocks LICM from hoisting the per-t re-reads of the
        // thresholds (global, L2-hot) and W fragments (LDS) into registers.
        int tz = 0;
        asm volatile("" : "+v"(tz));
        const float* inRowT = inRow + tz;

        const float* uRow = u + (size_t)t * tStride + (size_t)rowA * F_DIM;

        v8f acc[CT_COUNT];
#pragma unroll
        for (int ct = 0; ct < CT_COUNT; ++ct)
#pragma unroll
            for (int i = 0; i < 8; ++i) acc[ct][i] = bv[ct];

#pragma unroll
        for (int kc = 0; kc < KC_COUNT; ++kc) {
            const int kb = kc * 32 + koff;
            if (t + 1 < T_STEPS) {   // prefetch next time-step (global_prefetch_b8)
                __builtin_prefetch(uRow + tStride + kb, 0, 0);
            }
            // u: streamed once -> non-temporal; thresholds: cache-resident.
            v4f u0 = __builtin_nontemporal_load((const v4f*)(uRow + kb));
            v4f u1 = __builtin_nontemporal_load((const v4f*)(uRow + kb + 4));
            v4f u2 = __builtin_nontemporal_load((const v4f*)(uRow + kb + 16));
            v4f u3 = __builtin_nontemporal_load((const v4f*)(uRow + kb + 20));
            v4f r0 = *(const v4f*)(inRowT + kb);
            v4f r1 = *(const v4f*)(inRowT + kb + 4);
            v4f r2 = *(const v4f*)(inRowT + kb + 16);
            v4f r3 = *(const v4f*)(inRowT + kb + 20);

            v16h a;   // 16x32 f16 A fragment: spike = (u < rate)
#pragma unroll
            for (int i = 0; i < 4; ++i) {
                a[i]      = (u0[i] < r0[i]) ? (_Float16)1.0f : (_Float16)0.0f;
                a[4 + i]  = (u1[i] < r1[i]) ? (_Float16)1.0f : (_Float16)0.0f;
                a[8 + i]  = (u2[i] < r2[i]) ? (_Float16)1.0f : (_Float16)0.0f;
                a[12 + i] = (u3[i] < r3[i]) ? (_Float16)1.0f : (_Float16)0.0f;
            }

#pragma unroll
            for (int ct = 0; ct < CT_COUNT; ++ct) {
                const v16h bf = *(const v16h*)
                    &ldsW[(((kc * CT_COUNT + ct) * 32 + lane) << 4) + tz];
                acc[ct] = __builtin_amdgcn_wmma_f32_16x16x32_f16(
                    /*neg_a=*/false, a, /*neg_b=*/false, bf,
                    /*c_mod=*/(short)0, acc[ct],
                    /*reuse_a=*/false, /*reuse_b=*/false);
            }
        }

        // LIF euler step + hard spike (forward value of the surrogate) + soft reset
#pragma unroll
        for (int ct = 0; ct < CT_COUNT; ++ct)
#pragma unroll
            for (int i = 0; i < 8; ++i) {
                float vv = v[ct][i];
                vv += (acc[ct][i] - vv) * 0.1f;       // STEP_SIZE/TAU = 1/10
                float s = (vv > 1.0f) ? 1.0f : 0.0f;  // THRESHOLD = 1
                v[ct][i] = vv - s;                    // soft reset
                c[ct][i] += s;                        // spike count
            }
    }

    // Softmax over O=64 per row. Row M=j+8*(lane/16) lives in one 16-lane half:
    // 4 values per lane (one per column tile) -> shfl_xor masks 1,2,4,8.
#pragma unroll
    for (int j = 0; j < 8; ++j) {
        float m = fmaxf(fmaxf(c[0][j], c[1][j]), fmaxf(c[2][j], c[3][j]));
#pragma unroll
        for (int off = 1; off < 16; off <<= 1)
            m = fmaxf(m, __shfl_xor(m, off, 32));

        float e[CT_COUNT];
        float s = 0.0f;
#pragma unroll
        for (int ct = 0; ct < CT_COUNT; ++ct) {
            e[ct] = __expf(c[ct][j] - m);
            s += e[ct];
        }
#pragma unroll
        for (int off = 1; off < 16; off <<= 1)
            s += __shfl_xor(s, off, 32);

        const float inv = 1.0f / s;
        const int orow = rowBase + j + ((lane >> 4) << 3);
#pragma unroll
        for (int ct = 0; ct < CT_COUNT; ++ct)
            out[(size_t)orow * O_DIM + ct * 16 + (lane & 15)] = e[ct] * inv;
    }
}

extern "C" void kernel_launch(void* const* d_in, const int* in_sizes, int n_in,
                              void* d_out, int out_size, void* d_ws, size_t ws_size,
                              hipStream_t stream) {
    (void)n_in; (void)out_size; (void)d_ws; (void)ws_size;
    const float* inp = (const float*)d_in[0];   // [N,F]
    const float* u   = (const float*)d_in[1];   // [T,N,F]
    const float* W   = (const float*)d_in[2];   // [F,O]
    const float* b   = (const float*)d_in[3];   // [O]
    float* out = (float*)d_out;                 // [N,O]

    const int N = in_sizes[0] / F_DIM;
    const int grid = (N + ROWS_PER_BLOCK - 1) / ROWS_PER_BLOCK;   // 256 for N=32768
    snn_lif_fused<<<grid, WAVES_PER_BLOCK * 32, 0, stream>>>(inp, u, W, b, out, N);
}
